// MultiHeadAttention_69312182223681
// MI455X (gfx1250) — matrix operand order repaired and checked by reading, compiled, not run on hardware
//
#include <hip/hip_runtime.h>

// ---------------------------------------------------------------------------
// MI455X (gfx1250) multi-head causal attention, bf16 WMMA pipeline.
//   x,W in f32 -> convert to bf16 -> QKV proj GEMMs -> flash attention
//   -> output proj GEMM (f32 out).  All matmuls: v_wmma_f32_16x16x32_bf16.
// ---------------------------------------------------------------------------

typedef __bf16 bf16_t;
typedef __attribute__((ext_vector_type(16))) __bf16 v16bf;
typedef __attribute__((ext_vector_type(4)))  __bf16 v4bf;
typedef __attribute__((ext_vector_type(8)))  float  v8f;
typedef __attribute__((ext_vector_type(4)))  float  v4f;

union FragAB { v16bf v; v4f q[2]; };

constexpr int D_MODEL = 1024;
constexpr int N_HEADS = 16;
constexpr int D_K     = 64;
constexpr int T_SEQ   = 2048;
constexpr int BATCH   = 4;
constexpr int M_ROWS  = BATCH * T_SEQ;       // 8192
constexpr float LOG2E = 1.4426950408889634f;

#define NEG_INF (-__builtin_inff())

// ---------------------------------------------------------------------------
// 1) f32 -> bf16 elementwise convert (4 elems / thread, b128 load, b64 store)
// ---------------------------------------------------------------------------
__global__ __launch_bounds__(256, 4)
void cvt_bf16_kernel(const float* __restrict__ in, bf16_t* __restrict__ out)
{
    size_t i = ((size_t)blockIdx.x * 256 + threadIdx.x) * 4;
    v4f v = *(const v4f*)(in + i);
    v4bf o;
#pragma unroll
    for (int e = 0; e < 4; ++e) o[e] = (bf16_t)v[e];
    *(v4bf*)(out + i) = o;
}

// ---------------------------------------------------------------------------
// 2) W (K x N, f32, row-major) -> WT (N x K, bf16) via LDS 32x32 tile
// ---------------------------------------------------------------------------
__global__ __launch_bounds__(256, 2)
void transpose_cvt_kernel(const float* __restrict__ W, bf16_t* __restrict__ WT)
{
    __shared__ float tile[32][33];
    const int tx = threadIdx.x, ty = threadIdx.y;
    const int bx = blockIdx.x * 32;   // n base
    const int by = blockIdx.y * 32;   // k base
#pragma unroll
    for (int i = 0; i < 4; ++i)
        tile[ty + i * 8][tx] = W[(size_t)(by + ty + i * 8) * D_MODEL + bx + tx];
    __syncthreads();
#pragma unroll
    for (int i = 0; i < 4; ++i)
        WT[(size_t)(bx + ty + i * 8) * D_MODEL + by + tx] = (bf16_t)tile[tx][ty + i * 8];
}

// ---------------------------------------------------------------------------
// 3) GEMM  C(M=8192,N=1024) = A(M,K=1024) * BT(N,K)^T      (bf16 in, f32 acc)
//    MODE 0: out bf16, head-split layout (B,H,T,64), scaled (used for Q/K/V)
//    MODE 1: out f32 row-major (final projection)
//    WG = 256 thr (8 waves), tile 128x256, K-step 32, DOUBLE-buffered LDS
//    (one barrier per K step; global loads issued ahead of the WMMA block).
//    Wave tile 64x64 -> 16 f32 accumulators, 16 WMMA per K step,
//    16 ds_load_b128 per 16 WMMA (1.0 loads/WMMA vs 1.5 before).
// ---------------------------------------------------------------------------
template <int MODE>
__global__ __launch_bounds__(256, 1)
void gemm_bf16_wmma(const bf16_t* __restrict__ A, const bf16_t* __restrict__ BT,
                    void* __restrict__ Cout, float scale)
{
    constexpr int K   = D_MODEL;
    constexpr int LDT = 40;                 // padded LDS stride (halves), 80B
    const int tid  = threadIdx.x;
    const int lane = tid & 31, wave = tid >> 5;
    const int lh   = lane >> 4, ln = lane & 15;
    const int m0   = blockIdx.y * 128, n0 = blockIdx.x * 256;
    const int wm   = (wave & 1) * 64;       // 2 waves over M
    const int wn   = (wave >> 1) * 64;      // 4 waves over N

    __shared__ bf16_t As[2][128 * LDT];     // 2 x 10 KB
    __shared__ bf16_t Bs[2][256 * LDT];     // 2 x 20 KB   (total 60 KB)

    // Per-thread staging chunk coordinates (chunks of 8 halves).
    int ar[2], ak[2], br[4], bk[4];
#pragma unroll
    for (int i = 0; i < 2; ++i) { int c = tid + i * 256; ar[i] = c >> 2; ak[i] = (c & 3) * 8; }
#pragma unroll
    for (int i = 0; i < 4; ++i) { int c = tid + i * 256; br[i] = c >> 2; bk[i] = (c & 3) * 8; }

    v4f ra[2], rb[4];
    auto stage_regs = [&](int k0) {
#pragma unroll
        for (int i = 0; i < 2; ++i)
            ra[i] = *(const v4f*)&A[(size_t)(m0 + ar[i]) * K + k0 + ak[i]];
#pragma unroll
        for (int i = 0; i < 4; ++i)
            rb[i] = *(const v4f*)&BT[(size_t)(n0 + br[i]) * K + k0 + bk[i]];
    };
    auto store_lds = [&](int buf) {
#pragma unroll
        for (int i = 0; i < 2; ++i) *(v4f*)&As[buf][ar[i] * LDT + ak[i]] = ra[i];
#pragma unroll
        for (int i = 0; i < 4; ++i) *(v4f*)&Bs[buf][br[i] * LDT + bk[i]] = rb[i];
    };

    v8f acc[4][4] = {};

    stage_regs(0);
    store_lds(0);
    __syncthreads();

    int cur = 0;
    for (int k0 = 0; k0 < K; k0 += 32) {
        const bool has_next = (k0 + 32) < K;
        if (has_next) stage_regs(k0 + 32);          // hide global latency

        FragAB a[4];
#pragma unroll
        for (int mt = 0; mt < 4; ++mt) {            // A frag: K=0..7,16..23 (lanes<16)
            const bf16_t* p = &As[cur][(wm + mt * 16 + ln) * LDT + lh * 8];
            a[mt].q[0] = *(const v4f*)p;
            a[mt].q[1] = *(const v4f*)(p + 16);
        }
#pragma unroll
        for (int nt = 0; nt < 4; ++nt) {            // B frag: K=0..7,16..23 (lanes<16)
            FragAB b;
            const bf16_t* p = &Bs[cur][(wn + nt * 16 + ln) * LDT + lh * 8];
            b.q[0] = *(const v4f*)p;
            b.q[1] = *(const v4f*)(p + 16);
#pragma unroll
            for (int mt = 0; mt < 4; ++mt)
                acc[mt][nt] = __builtin_amdgcn_wmma_f32_16x16x32_bf16(
                    false, a[mt].v, false, b.v, (short)0, acc[mt][nt], false, false);
        }

        if (has_next) store_lds(cur ^ 1);           // fill other buffer
        __syncthreads();
        cur ^= 1;
    }

#pragma unroll
    for (int mt = 0; mt < 4; ++mt)
#pragma unroll
        for (int nt = 0; nt < 4; ++nt)
#pragma unroll
            for (int r = 0; r < 8; ++r) {
                int row = m0 + wm + mt * 16 + r + 8 * lh;
                int col = n0 + wn + nt * 16 + ln;
                float v = acc[mt][nt][r] * scale;
                if (MODE == 0) {
                    // (b, h, t, d) = (row>>11, col>>6, row&2047, col&63)
                    size_t idx = ((((size_t)(row >> 11) * N_HEADS + (col >> 6)) * T_SEQ)
                                  + (row & (T_SEQ - 1))) * D_K + (col & (D_K - 1));
                    ((bf16_t*)Cout)[idx] = (bf16_t)v;
                } else {
                    ((float*)Cout)[(size_t)row * D_MODEL + col] = v;
                }
            }
}

// ---------------------------------------------------------------------------
// 4) Flash attention, causal.  Grid (T/128, B*H).  WG = 8 waves; each wave
//    owns 16 query rows.  Bc = 64 keys per iteration.
//    S = Q K^T (WMMA), online softmax (f32, exp2 domain — log2e folded into
//    Q at projection), P -> wave-private LDS -> A-frags, O += P V (WMMA).
// ---------------------------------------------------------------------------
__global__ __launch_bounds__(256, 1)
void flash_attn_kernel(const bf16_t* __restrict__ Q, const bf16_t* __restrict__ Kg,
                       const bf16_t* __restrict__ Vg, bf16_t* __restrict__ ctx)
{
    const int tid  = threadIdx.x;
    const int lane = tid & 31, wave = tid >> 5;
    const int lh   = lane >> 4, ln = lane & 15;
    const int q0   = blockIdx.x * 128;
    const int bh   = blockIdx.y;              // b*16 + h
    const int b    = bh >> 4, h = bh & 15;

    __shared__ bf16_t Kt[64 * 72];            // K block row-major (j, d)
    __shared__ bf16_t Vt[64 * 72];            // V block transposed (d, j)
    __shared__ bf16_t Pl[8 * 16 * 72];        // per-wave P scratch
    bf16_t* Pw = &Pl[wave * 16 * 72];

    // Preload Q A-fragments for this wave's 16 rows (d = 0..63 -> 2 frags).
    FragAB qa[2];
    {
        const bf16_t* Qrow = Q + ((size_t)bh * T_SEQ + q0 + wave * 16 + ln) * D_K;
#pragma unroll
        for (int f = 0; f < 2; ++f) {
            int d0 = f * 32 + lh * 8;
            qa[f].q[0] = *(const v4f*)(Qrow + d0);
            qa[f].q[1] = *(const v4f*)(Qrow + d0 + 16);
        }
    }

    v8f   O[4] = {};
    float mrow[8], lrow[8];
#pragma unroll
    for (int r = 0; r < 8; ++r) { mrow[r] = NEG_INF; lrow[r] = 0.f; }

    const bf16_t* Kbase = Kg + (size_t)bh * T_SEQ * D_K;
    const bf16_t* Vbase = Vg + (size_t)bh * T_SEQ * D_K;
    const int iMin = q0 + wave * 16;

    for (int s0 = 0; s0 <= q0 + 64; s0 += 64) {
        __syncthreads();
        // Stage K (row-major) and V (transposed) into LDS.
#pragma unroll
        for (int i = 0; i < 2; ++i) {
            int c  = tid + i * 256;           // 512 chunks of 8 halves
            int jr = c >> 3, kc = (c & 7) * 8;
            *(v4f*)&Kt[jr * 72 + kc] = *(const v4f*)&Kbase[(size_t)(s0 + jr) * D_K + kc];
            v4f tmp = *(const v4f*)&Vbase[(size_t)(s0 + jr) * D_K + kc];
            const bf16_t* tp = (const bf16_t*)&tmp;
#pragma unroll
            for (int e = 0; e < 8; ++e) Vt[(kc + e) * 72 + jr] = tp[e];
        }
        if (s0 < q0 + 64) {                   // prefetch next K/V block
            __builtin_prefetch(Kbase + (size_t)(s0 + 64) * D_K + tid * 16, 0, 1);
            __builtin_prefetch(Vbase + (size_t)(s0 + 64) * D_K + tid * 16, 0, 1);
        }
        __syncthreads();

        // S = Q * K^T  (16 x 64 per wave)
        v8f S[4] = {};
#pragma unroll
        for (int kk = 0; kk < 2; ++kk)
#pragma unroll
            for (int nt = 0; nt < 4; ++nt) {
                FragAB bf;
                const bf16_t* kp = &Kt[(nt * 16 + ln) * 72 + kk * 32 + lh * 8];
                bf.q[0] = *(const v4f*)kp;
                bf.q[1] = *(const v4f*)(kp + 16);
                S[nt] = __builtin_amdgcn_wmma_f32_16x16x32_bf16(
                    false, qa[kk].v, false, bf.v, (short)0, S[nt], false, false);
            }

        // Causal mask (only on diagonal-crossing blocks for this wave).
        if (s0 + 63 > iMin) {
#pragma unroll
            for (int nt = 0; nt < 4; ++nt) {
                int j = s0 + nt * 16 + ln;
#pragma unroll
                for (int r = 0; r < 8; ++r)
                    if (j > iMin + r + 8 * lh) S[nt][r] = NEG_INF;
            }
        }

        // Online softmax (exp2 domain; scale*log2e already folded into Q).
        float alpha[8];
#pragma unroll
        for (int r = 0; r < 8; ++r) {
            float rm = fmaxf(fmaxf(S[0][r], S[1][r]), fmaxf(S[2][r], S[3][r]));
            rm = fmaxf(rm, __shfl_xor(rm, 1, 32));
            rm = fmaxf(rm, __shfl_xor(rm, 2, 32));
            rm = fmaxf(rm, __shfl_xor(rm, 4, 32));
            rm = fmaxf(rm, __shfl_xor(rm, 8, 32));
            float mnew = fmaxf(mrow[r], rm);
            alpha[r] = __builtin_amdgcn_exp2f(mrow[r] - mnew);
            mrow[r]  = mnew;
            float rs = 0.f;
#pragma unroll
            for (int nt = 0; nt < 4; ++nt) {
                float p = __builtin_amdgcn_exp2f(S[nt][r] - mnew);
                S[nt][r] = p;
                rs += p;
            }
            rs += __shfl_xor(rs, 1, 32);
            rs += __shfl_xor(rs, 2, 32);
            rs += __shfl_xor(rs, 4, 32);
            rs += __shfl_xor(rs, 8, 32);
            lrow[r] = lrow[r] * alpha[r] + rs;
#pragma unroll
            for (int dt = 0; dt < 4; ++dt) O[dt][r] *= alpha[r];
        }

        // P (C-layout) -> wave-private LDS (A-layout re-read below).
#pragma unroll
        for (int nt = 0; nt < 4; ++nt)
#pragma unroll
            for (int r = 0; r < 8; ++r)
                Pw[(r + 8 * lh) * 72 + nt * 16 + ln] = (bf16_t)S[nt][r];

        // O += P * V   (V staged transposed: rows = d, cols = key)
#pragma unroll
        for (int kk = 0; kk < 2; ++kk) {
            FragAB af;
            const bf16_t* pp = &Pw[ln * 72 + kk * 32 + lh * 8];
            af.q[0] = *(const v4f*)pp;
            af.q[1] = *(const v4f*)(pp + 16);
#pragma unroll
            for (int dt = 0; dt < 4; ++dt) {
                FragAB bf;
                const bf16_t* vp = &Vt[(dt * 16 + ln) * 72 + kk * 32 + lh * 8];
                bf.q[0] = *(const v4f*)vp;
                bf.q[1] = *(const v4f*)(vp + 16);
                O[dt] = __builtin_amdgcn_wmma_f32_16x16x32_bf16(
                    false, af.v, false, bf.v, (short)0, O[dt], false, false);
            }
        }
    }

    // Normalize and write context (row-major 8192 x 1024, col = h*64 + d).
#pragma unroll
    for (int r = 0; r < 8; ++r) {
        float rl = __builtin_amdgcn_rcpf(lrow[r]);
        int t = q0 + wave * 16 + r + 8 * lh;
#pragma unroll
        for (int dt = 0; dt < 4; ++dt) {
            int col = h * D_K + dt * 16 + ln;
            ctx[((size_t)b * T_SEQ + t) * D_MODEL + col] = (bf16_t)(O[dt][r] * rl);
        }
    }
}

// ---------------------------------------------------------------------------
// Orchestration
// ---------------------------------------------------------------------------
extern "C" void kernel_launch(void* const* d_in, const int* in_sizes, int n_in,
                              void* d_out, int out_size, void* d_ws, size_t ws_size,
                              hipStream_t stream)
{
    (void)in_sizes; (void)n_in; (void)out_size; (void)ws_size;

    const float* x  = (const float*)d_in[0];
    const float* Wq = (const float*)d_in[1];
    const float* Wk = (const float*)d_in[2];
    const float* Wv = (const float*)d_in[3];
    const float* Wo = (const float*)d_in[4];
    float* out = (float*)d_out;

    char* w = (char*)d_ws;
    const size_t MB = 1024 * 1024;
    bf16_t* xh  = (bf16_t*)(w);              // 16 MB: x in bf16
    bf16_t* WqT = (bf16_t*)(w + 16 * MB);    //  2 MB each, transposed (N,K)
    bf16_t* WkT = (bf16_t*)(w + 18 * MB);
    bf16_t* WvT = (bf16_t*)(w + 20 * MB);
    bf16_t* WoT = (bf16_t*)(w + 22 * MB);
    bf16_t* Qb  = (bf16_t*)(w + 24 * MB);    // 16 MB, (B,H,T,64)
    bf16_t* Kb  = (bf16_t*)(w + 40 * MB);
    bf16_t* Vb  = (bf16_t*)(w + 56 * MB);
    bf16_t* Ctx = (bf16_t*)(w + 72 * MB);    // 16 MB, (8192,1024)

    // 1) converts
    cvt_bf16_kernel<<<(M_ROWS * D_MODEL) / 1024, 256, 0, stream>>>(x, xh);
    dim3 tg(D_MODEL / 32, D_MODEL / 32), tb(32, 8);
    transpose_cvt_kernel<<<tg, tb, 0, stream>>>(Wq, WqT);
    transpose_cvt_kernel<<<tg, tb, 0, stream>>>(Wk, WkT);
    transpose_cvt_kernel<<<tg, tb, 0, stream>>>(Wv, WvT);
    transpose_cvt_kernel<<<tg, tb, 0, stream>>>(Wo, WoT);

    // 2) Q/K/V projections (Q carries softmax scale * log2e)
    dim3 gg(D_MODEL / 256, M_ROWS / 128);
    gemm_bf16_wmma<0><<<gg, 256, 0, stream>>>(xh, WqT, Qb, 0.125f * LOG2E);
    gemm_bf16_wmma<0><<<gg, 256, 0, stream>>>(xh, WkT, Kb, 1.0f);
    gemm_bf16_wmma<0><<<gg, 256, 0, stream>>>(xh, WvT, Vb, 1.0f);

    // 3) causal flash attention
    flash_attn_kernel<<<dim3(T_SEQ / 128, BATCH * N_HEADS), 256, 0, stream>>>(
        Qb, Kb, Vb, Ctx);

    // 4) output projection -> f32
    gemm_bf16_wmma<1><<<gg, 256, 0, stream>>>(Ctx, WoT, out, 1.0f);
}
